// GraphAttentionLayer_54571854463366
// MI455X (gfx1250) — matrix operand order repaired and checked by reading, not compiled, not run on hardware
//
#include <hip/hip_runtime.h>
#include <hip/hip_bf16.h>
#include <math.h>

typedef __bf16 bf16;
typedef __attribute__((ext_vector_type(16))) __bf16 v16bf;
typedef __attribute__((ext_vector_type(8)))  float  v8f;
typedef __attribute__((ext_vector_type(4)))  int    v4i;
typedef __attribute__((ext_vector_type(4)))  unsigned int u32x4;
typedef __attribute__((ext_vector_type(8)))  int    i32x8;
typedef __attribute__((ext_vector_type(4)))  int    i32x4;

union Frag {
    v16bf bf;
    v4i   i4[2];
    bf16  h[16];
};

constexpr int   kN    = 2048;
constexpr int   kC    = 256;
constexpr int   kEMB  = 768;   // 3*C
constexpr int   kH    = 8;
constexpr int   kD    = 96;    // EMB/H
constexpr float kALPHA = 0.2f;
constexpr float kSCALE = 0.10206207261596575f;  // 96^-0.5

#if defined(__gfx1250__) && __has_builtin(__builtin_amdgcn_tensor_load_to_lds) && \
    __has_builtin(__builtin_amdgcn_s_wait_tensorcnt)
#define HAS_TDM 1
#else
#define HAS_TDM 0
#endif

#if defined(__gfx1250__) && !HAS_TDM
# warning "gfx1250 device pass: TDM builtin NOT available, using global-load fallback"
#endif

// ---------------------------------------------------------------------------
// f32 -> bf16 convert
// ---------------------------------------------------------------------------
__global__ void cvt_f32_bf16(const float* __restrict__ src, bf16* __restrict__ dst, int n) {
    int i = blockIdx.x * blockDim.x + threadIdx.x;
    if (i < n) dst[i] = (bf16)src[i];
}

// ---------------------------------------------------------------------------
// e = concat(eh, er, et) -> bf16  (row i has 768 elements)
// ---------------------------------------------------------------------------
__global__ void concat_e(const float* __restrict__ eh, const float* __restrict__ er,
                         const float* __restrict__ et, bf16* __restrict__ e) {
    int idx = blockIdx.x * blockDim.x + threadIdx.x;
    if (idx >= kN * kEMB) return;
    int i = idx / kEMB;
    int c = idx % kEMB;
    float v;
    if (c < kC)            v = eh[i * kC + c];
    else if (c < 2 * kC)   v = er[i * kC + (c - kC)];
    else                   v = et[i * kC + (c - 2 * kC)];
    e[idx] = (bf16)v;
}

// ---------------------------------------------------------------------------
// WMMA GEMM:  C[m][n] = sum_k A[m][k] * W[n][k]  (+ bias[n]) (+ addend[m][n])
// A: M x K bf16 row-major;  W: N x K bf16 row-major (i.e. B = W^T)
// Block = 128 threads = 4 waves; wave w -> 16x64 tile (4 accumulators, A reused 4x)
// K-loop is a copy-free even/odd ping-pong (requires K % 64 == 0): loads for
// chunk k+32 are issued into the idle register set while chunk k's 4 WMMAs run.
// ---------------------------------------------------------------------------
__global__ __launch_bounds__(128) void wmma_gemm_bf16(
    const bf16* __restrict__ A, const bf16* __restrict__ W,
    const float* __restrict__ bias, const float* __restrict__ addend,
    float* __restrict__ outF, bf16* __restrict__ outB,
    int M, int N, int K)
{
    const int lane = threadIdx.x & 31;
    const int wave = threadIdx.x >> 5;
    const int l16  = lane & 15;
    const int hl   = lane >> 4;           // 0 | 1
    const int m0   = blockIdx.x * 16;
    const int n0   = blockIdx.y * 256 + wave * 64;

    const bf16* Arow = A + (size_t)(m0 + l16) * K;
    const bf16* Wr[4];
#pragma unroll
    for (int nt = 0; nt < 4; ++nt)
        Wr[nt] = W + (size_t)(n0 + nt * 16 + l16) * K + 16 * hl;

    v8f acc[4];
#pragma unroll
    for (int nt = 0; nt < 4; ++nt) acc[nt] = (v8f){0.f, 0.f, 0.f, 0.f, 0.f, 0.f, 0.f, 0.f};

    auto loadA = [&](Frag& f, int k0) {
        const v4i* ap = (const v4i*)(Arow + k0);
        f.i4[0] = ap[2 * hl];        // lane<16: dwords {0..3}; lane>=16: {8..11}
        f.i4[1] = ap[2 * hl + 1];    // lane<16: dwords {4..7}; lane>=16: {12..15}
    };
    auto loadB4 = [&](Frag* f4, int k0) {
#pragma unroll
        for (int nt = 0; nt < 4; ++nt) {
            const v4i* bp = (const v4i*)(Wr[nt] + k0);
            f4[nt].i4[0] = bp[0];
            f4[nt].i4[1] = bp[1];
        }
    };
    auto wmma4 = [&](const Frag& af, const Frag* b4) {
        acc[0] = __builtin_amdgcn_wmma_f32_16x16x32_bf16(false, af.bf, false, b4[0].bf,
                                                         (short)0, acc[0], false, false);
        acc[1] = __builtin_amdgcn_wmma_f32_16x16x32_bf16(false, af.bf, false, b4[1].bf,
                                                         (short)0, acc[1], false, false);
        acc[2] = __builtin_amdgcn_wmma_f32_16x16x32_bf16(false, af.bf, false, b4[2].bf,
                                                         (short)0, acc[2], false, false);
        acc[3] = __builtin_amdgcn_wmma_f32_16x16x32_bf16(false, af.bf, false, b4[3].bf,
                                                         (short)0, acc[3], false, false);
    };

    Frag afA, bA[4], afB, bB[4];
    loadA(afA, 0);
    loadB4(bA, 0);

#pragma unroll 1
    for (int k0 = 0; k0 < K; k0 += 64) {
        // fill the odd set while the even set computes
        if (k0 + 32 < K) {
            __builtin_prefetch(Arow + k0 + 64, 0, 3);   // -> global_prefetch_b8
            loadA(afB, k0 + 32);
            loadB4(bB, k0 + 32);
        }
        wmma4(afA, bA);
        // refill the even set while the odd set computes
        if (k0 + 64 < K) {
            loadA(afA, k0 + 64);
            loadB4(bA, k0 + 64);
        }
        wmma4(afB, bB);
    }

#pragma unroll
    for (int nt = 0; nt < 4; ++nt) {
        const int col = n0 + nt * 16 + l16;
        const float b = bias ? bias[col] : 0.f;
#pragma unroll
        for (int r = 0; r < 8; ++r) {
            int row = m0 + r + 8 * hl;
            float v = acc[nt][r] + b;
            if (addend) v += addend[(size_t)row * N + col];
            if (outF) outF[(size_t)row * N + col] = v;
            if (outB) outB[(size_t)row * N + col] = (bf16)v;
        }
    }
}

// ---------------------------------------------------------------------------
// s1[i,h] = Wh[i,h,:] . a[:96] ;  s2[i,h] = Wh[i,h,:] . a[96:192]
// ---------------------------------------------------------------------------
__global__ void s1s2_kernel(const float* __restrict__ Wh, const float* __restrict__ a,
                            float* __restrict__ s1, float* __restrict__ s2) {
    int t = blockIdx.x * blockDim.x + threadIdx.x;
    if (t >= kN * kH) return;
    int i = t >> 3, h = t & 7;
    const float* w = Wh + (size_t)i * kEMB + h * kD;
    float x1 = 0.f, x2 = 0.f;
#pragma unroll 8
    for (int d = 0; d < kD; ++d) {
        float wv = w[d];
        x1 += wv * a[d];
        x2 += wv * a[kD + d];
    }
    s1[t] = x1;
    s2[t] = x2;
}

// ---------------------------------------------------------------------------
// Two-pass softmax stats over j for each (i, h)
// ---------------------------------------------------------------------------
__global__ void softmax_stats(const float* __restrict__ s1, const float* __restrict__ s2,
                              const int* __restrict__ h_id,
                              float* __restrict__ mOut, float* __restrict__ lOut) {
    int t = blockIdx.x * blockDim.x + threadIdx.x;
    if (t >= kN * kH) return;
    int i = t >> 3, h = t & 7;
    int hidi = h_id[i];
    float s1r[8];
#pragma unroll
    for (int q = 0; q < 8; ++q) s1r[q] = s1[i * kH + q];

    float m = -INFINITY;
    for (int j = 0; j < kN; ++j) {
        if (h_id[j] != hidi) continue;
        int   q = ((j << 3) + h) >> 11;
        float x = s1r[q] + s2[j * kH + h];
        float z = (x >= 0.f ? x : kALPHA * x) * kSCALE;
        m = fmaxf(m, z);
    }
    float l = 0.f;
    for (int j = 0; j < kN; ++j) {
        if (h_id[j] != hidi) continue;
        int   q = ((j << 3) + h) >> 11;
        float x = s1r[q] + s2[j * kH + h];
        float z = (x >= 0.f ? x : kALPHA * x) * kSCALE;
        l += __expf(z - m);
    }
    mOut[t] = m;
    lOut[t] = l;
}

// ---------------------------------------------------------------------------
// Attention apply: out[i,h,:] = sum_j softmax_j(z) * Wh[j,h,:]
// One wave per (16-row i-tile, head). Wh chunk staged via the Tensor Data
// Mover (TDM) into LDS when available, then transposed for B fragments.
// ---------------------------------------------------------------------------
__global__ __launch_bounds__(32) void attention_wmma(
    const float* __restrict__ s1, const float* __restrict__ s2,
    const int* __restrict__ h_id,
    const float* __restrict__ mIn, const float* __restrict__ lIn,
    const bf16* __restrict__ WhB, bf16* __restrict__ outB)
{
    __shared__ bf16  ldsRow[32 * kD];  // [j][d] staging tile (TDM target), 6 KB
    __shared__ bf16  ldsT[kD * 32];    // [d][j] transposed, 6 KB
    __shared__ float s2sh[32];
    __shared__ int   hsh[32];

    const int lane = threadIdx.x;
    const int l16  = lane & 15;
    const int hl   = lane >> 4;
    const int i0   = blockIdx.x * 16;
    const int h    = blockIdx.y;

    const int   rowi = i0 + l16;
    const float mi   = mIn[rowi * kH + h];
    const float rli  = 1.0f / lIn[rowi * kH + h];
    const int   hidi = h_id[rowi];
    float s1r[8];
#pragma unroll
    for (int q = 0; q < 8; ++q) s1r[q] = s1[rowi * kH + q];

    v8f acc[6];
#pragma unroll
    for (int nt = 0; nt < 6; ++nt) acc[nt] = (v8f){0.f, 0.f, 0.f, 0.f, 0.f, 0.f, 0.f, 0.f};

#if HAS_TDM
    const unsigned int ldsBase = (unsigned int)(uintptr_t)(void*)&ldsRow[0];
#endif

    for (int j0 = 0; j0 < kN; j0 += 32) {
#if HAS_TDM
        {
            // Tensor DMA: load 96(x) x 32(y) bf16 tile, dim0 stride 768 elems,
            // from WhB[j0][h*96] into ldsRow (row-major [j][d]).
            unsigned long long ga =
                (unsigned long long)(uintptr_t)(WhB + (size_t)j0 * kEMB + h * kD);
            u32x4 g0;
            g0[0] = 1u;                                   // count=1, user descriptor
            g0[1] = ldsBase;                              // lds_addr (bytes)
            g0[2] = (unsigned int)(ga & 0xffffffffu);     // global_addr[31:0]
            g0[3] = (unsigned int)((ga >> 32) & 0x1ffffffu) | (2u << 30);  // [56:32] | type=2
            i32x8 g1;
            g1[0] = (int)(1u << 16);                      // workgroup_mask=0, data_size=1 (2B)
            g1[1] = (int)((unsigned)kD << 16);            // tensor_dim0 = 96 (bits 79:48 lo)
            g1[2] = (int)(32u << 16);                     // tensor_dim1 = 32 (bits 111:80 lo)
            g1[3] = (int)((unsigned)kD << 16);            // tile_dim0 = 96 (bits 127:112)
            g1[4] = 32;                                   // tile_dim1 = 32 (bits 143:128)
            g1[5] = kEMB;                                 // tensor_dim0_stride = 768
            g1[6] = 0;
            g1[7] = 0;
            i32x4 gz = {0, 0, 0, 0};
#if __clang_major__ >= 23
            i32x8 gz8 = {0, 0, 0, 0, 0, 0, 0, 0};
            __builtin_amdgcn_tensor_load_to_lds(g0, g1, gz, gz, gz8, 0);
#else
            __builtin_amdgcn_tensor_load_to_lds(g0, g1, gz, gz, 0);
#endif
            __builtin_amdgcn_s_wait_tensorcnt(0);
        }
        __syncthreads();
        const v4i* src = (const v4i*)(ldsRow + lane * kD);
#else
        const v4i* src = (const v4i*)(WhB + (size_t)(j0 + lane) * kEMB + h * kD);
#endif
        // --- transpose 32x24 (d < 24) part of the tile into ldsT[d][j] ---
#pragma unroll
        for (int c = 0; c < 3; ++c) {       // 24 bf16 = 3 x 16B
            v4i v = src[c];
            const bf16* hv = (const bf16*)&v;
#pragma unroll
            for (int e = 0; e < 8; ++e)
                ldsT[(c * 8 + e) * 32 + lane] = hv[e];
        }
        s2sh[lane] = s2[(j0 + lane) * kH + h];
        hsh[lane]  = h_id[j0 + lane];
        __syncthreads();

        // --- synthesize probability tile directly in A-fragment layout ---
        Frag af;
#pragma unroll
        for (int t = 0; t < 16; ++t) {
            int v    = t >> 1;
            int koff = 16 * hl + t; (void)v;
            float p;
            if (hsh[koff] != hidi) {
                p = 0.f;
            } else {
                int   j = j0 + koff;
                int   q = ((j << 3) + h) >> 11;
                float x = s1r[q] + s2sh[koff];
                float z = (x >= 0.f ? x : kALPHA * x) * kSCALE;
                p = __expf(z - mi) * rli;
            }
            af.h[t] = (bf16)p;
        }

        // --- 6 WMMAs: P(16x32) x Wh(32x16) per 16-wide d-tile ---
#pragma unroll
        for (int nt = 0; nt < 6; ++nt) {
            Frag bfv;
            const v4i* bp = (const v4i*)&ldsT[(nt * 16 + l16) * 32 + 16 * hl];
            bfv.i4[0] = bp[0];
            bfv.i4[1] = bp[1];
            acc[nt] = __builtin_amdgcn_wmma_f32_16x16x32_bf16(false, af.bf, false, bfv.bf,
                                                              (short)0, acc[nt], false, false);
        }
        __syncthreads();
    }

    // --- store 16x96 tile as bf16 (consumed by proj GEMM) ---
#pragma unroll
    for (int nt = 0; nt < 6; ++nt) {
#pragma unroll
        for (int r = 0; r < 8; ++r) {
            int row = i0 + r + 8 * hl;
            int col = h * kD + nt * 16 + l16;
            outB[(size_t)row * kEMB + col] = (bf16)acc[nt][r];
        }
    }
}

// ---------------------------------------------------------------------------
// Launch
// ---------------------------------------------------------------------------
extern "C" void kernel_launch(void* const* d_in, const int* in_sizes, int n_in,
                              void* d_out, int out_size, void* d_ws, size_t ws_size,
                              hipStream_t stream) {
    const float* eh     = (const float*)d_in[0];
    const float* er     = (const float*)d_in[1];
    const float* et     = (const float*)d_in[2];
    const int*   h_id   = (const int*)  d_in[3];
    const float* W_w    = (const float*)d_in[4];
    const float* a      = (const float*)d_in[5];
    const float* proj_w = (const float*)d_in[6];
    const float* proj_b = (const float*)d_in[7];
    const float* skip_w = (const float*)d_in[8];
    const float* skip_b = (const float*)d_in[9];
    float* out = (float*)d_out;

    // workspace carve (all 256B aligned)
    char* ws = (char*)d_ws;
    auto carve = [&](size_t bytes) { char* p = ws; ws += (bytes + 255) & ~(size_t)255; return p; };
    bf16*  eB      = (bf16*) carve((size_t)kN * kEMB * 2);
    bf16*  WwB     = (bf16*) carve((size_t)kEMB * kEMB * 2);
    bf16*  projB   = (bf16*) carve((size_t)kEMB * kEMB * 2);
    bf16*  skipB   = (bf16*) carve((size_t)kEMB * kEMB * 2);
    float* WhF     = (float*)carve((size_t)kN * kEMB * 4);
    bf16*  WhB     = (bf16*) carve((size_t)kN * kEMB * 2);
    float* s1      = (float*)carve((size_t)kN * kH * 4);
    float* s2      = (float*)carve((size_t)kN * kH * 4);
    float* mBuf    = (float*)carve((size_t)kN * kH * 4);
    float* lBuf    = (float*)carve((size_t)kN * kH * 4);
    bf16*  attnB   = (bf16*) carve((size_t)kN * kEMB * 2);

    const int nW = kEMB * kEMB;
    cvt_f32_bf16<<<(nW + 255) / 256, 256, 0, stream>>>(W_w,    WwB,   nW);
    cvt_f32_bf16<<<(nW + 255) / 256, 256, 0, stream>>>(proj_w, projB, nW);
    cvt_f32_bf16<<<(nW + 255) / 256, 256, 0, stream>>>(skip_w, skipB, nW);
    concat_e<<<(kN * kEMB + 255) / 256, 256, 0, stream>>>(eh, er, et, eB);

    dim3 gGemm(kN / 16, kEMB / 256);
    // Wh = e @ W_w^T  (f32 + bf16 copies)
    wmma_gemm_bf16<<<gGemm, 128, 0, stream>>>(eB, WwB, nullptr, nullptr, WhF, WhB,
                                              kN, kEMB, kEMB);
    s1s2_kernel<<<(kN * kH + 255) / 256, 256, 0, stream>>>(WhF, a, s1, s2);
    softmax_stats<<<(kN * kH + 255) / 256, 256, 0, stream>>>(s1, s2, h_id, mBuf, lBuf);

    attention_wmma<<<dim3(kN / 16, kH), 32, 0, stream>>>(s1, s2, h_id, mBuf, lBuf, WhB, attnB);

    // out = attn_out @ proj_w^T + proj_b
    wmma_gemm_bf16<<<gGemm, 128, 0, stream>>>(attnB, projB, proj_b, nullptr, out, nullptr,
                                              kN, kEMB, kEMB);
    // out += e @ skip_w^T + skip_b
    wmma_gemm_bf16<<<gGemm, 128, 0, stream>>>(eB, skipB, skip_b, out, out, nullptr,
                                              kN, kEMB, kEMB);
}
